// TCN_DCM_32074815766748
// MI455X (gfx1250) — matrix operand order repaired and checked by reading, not compiled, not run on hardware
//
#include <hip/hip_runtime.h>

// ---------------------------------------------------------------------------
// MI455X (gfx1250) 2D-TAN forward. All GEMM-shaped work (1x1 convs, attention
// projections, QK^T, PV, implicit-GEMM 5x5 convs) -> V_WMMA_F32_16X16X32_BF16
// with f32 accumulation. ~1.45 TFLOP vs ~45us of HBM traffic @23.3TB/s =>
// matrix-compute bound; weights (~120MB) live in the 192MB L2.
// Fragment-friendly data layouts (position-major K/Q, transposed LDS B tiles,
// tap-major conv weights) so every WMMA operand is fetched with b128 vector
// loads, plus TDM (tensor_load_to_lds, 6-arg clang-23 builtin) for A tiles.
// ---------------------------------------------------------------------------

#define USE_TDM 1

static constexpr int BZC  = 8;
static constexpr int DIMC = 512;
static constexpr int NC   = 64;
static constexpr int NNC  = NC * NC;           // 4096
static constexpr int MC   = NC * (NC + 1) / 2; // 2080
static constexpr int HDC  = 128;               // head dim
static constexpr int HEADSC = 4;

typedef __attribute__((ext_vector_type(16))) __bf16 v16bf;
typedef __attribute__((ext_vector_type(8)))  float  v8f;

union FragU { v16bf v; float4 f[2]; };

__device__ __forceinline__ __bf16 f2bf(float f) {
  union { float f; unsigned u; } v; v.f = f;
  unsigned r = v.u + 0x7FFFu + ((v.u >> 16) & 1u);   // round-to-nearest-even
  unsigned short h = (unsigned short)(r >> 16);
  __bf16 out; __builtin_memcpy(&out, &h, 2); return out;
}
__device__ __forceinline__ float bf2f(__bf16 b) {
  unsigned short h; __builtin_memcpy(&h, &b, 2);
  union { unsigned u; float f; } v; v.u = ((unsigned)h) << 16; return v.f;
}

#if USE_TDM
// Build a D# (ISA 08_async_tensor sec 8.3-8.5) for a 2D row-major bf16 tile and
// issue TENSOR_LOAD_TO_LDS. 6-arg (clang-23 / therock-10.0) builtin form.
__device__ __forceinline__ void tdm_load_tile_2d(
    unsigned lds_off, const void* gaddr, int rowlen_elems, int nrows,
    int tile_w, int tile_h, int row_stride_elems) {
  typedef __attribute__((ext_vector_type(4))) unsigned tdm_u32x4;
  typedef __attribute__((ext_vector_type(8))) int tdm_i32x8;
  typedef __attribute__((ext_vector_type(4))) int tdm_i32x4;
  unsigned long long ga = (unsigned long long)(size_t)gaddr;
  tdm_u32x4 g0;
  g0[0] = 1u;                                            // count=1, user mode
  g0[1] = lds_off;                                       // lds_addr (bytes)
  g0[2] = (unsigned)ga;                                  // global_addr[31:0]
  g0[3] = (unsigned)((ga >> 32) & 0x01FFFFFFull) | (2u << 30);  // addr[56:32]|type=2
  tdm_i32x8 g1;
  g1[0] = (1 << 16);                                     // data_size=1 -> 2 bytes
  g1[1] = (int)((rowlen_elems & 0xFFFF) << 16);          // tensor_dim0 lo16 @63:48
  g1[2] = (int)(((unsigned)rowlen_elems >> 16) & 0xFFFFu)
        | (int)((nrows & 0xFFFF) << 16);                 // dim0 hi16 | dim1 lo16
  g1[3] = (int)(((unsigned)nrows >> 16) & 0xFFFFu)
        | (int)((tile_w & 0xFFFF) << 16);                // dim1 hi16 | tile_dim0
  g1[4] = (tile_h & 0xFFFF);                             // tile_dim1 (tile_dim2=0)
  g1[5] = row_stride_elems;                              // tensor_dim0_stride lo32
  g1[6] = 0; g1[7] = 0;
  tdm_i32x4 gz4 = {0, 0, 0, 0};
  tdm_i32x8 gz8 = {0, 0, 0, 0, 0, 0, 0, 0};
  __builtin_amdgcn_tensor_load_to_lds(g0, g1, gz4, gz4, gz8, 0);
}
#endif

// f32 -> bf16 converter (grid-stride)
__global__ void cvt_bf16(const float* __restrict__ src, __bf16* __restrict__ dst, size_t n) {
  size_t i = (size_t)blockIdx.x * blockDim.x + threadIdx.x;
  size_t st = (size_t)gridDim.x * blockDim.x;
  for (; i < n; i += st) dst[i] = f2bf(src[i]);
}

// repack 5x5 conv weights (O,I,5,5) f32 -> tap-major (tap, O, I) bf16
__global__ void repack_convw(const float* __restrict__ W, __bf16* __restrict__ Wt) {
  size_t total = (size_t)25 * DIMC * DIMC;
  size_t i = (size_t)blockIdx.x * blockDim.x + threadIdx.x;
  size_t st = (size_t)gridDim.x * blockDim.x;
  for (; i < total; i += st) {
    size_t tap = i / ((size_t)DIMC * DIMC);
    size_t rem = i % ((size_t)DIMC * DIMC);
    size_t oc = rem / DIMC, ci = rem % DIMC;
    Wt[i] = f2bf(W[(oc * DIMC + ci) * 25 + tap]);
  }
}

// qQ = relu(qry)@W_Q^T + b_Q ; q2 = qry@W_fc^T + b_fc  (tiny: 8x512 each)
__global__ void qvec_kernel(const float* __restrict__ qry,
                            const float* __restrict__ W_Q, const float* __restrict__ b_Q,
                            const float* __restrict__ W_fc, const float* __restrict__ b_fc,
                            float* __restrict__ qQ, float* __restrict__ q2) {
  int b = blockIdx.x, o = threadIdx.x;
  float s1 = 0.f, s2 = 0.f;
  for (int i = 0; i < DIMC; ++i) {
    float q = qry[b * DIMC + i];
    s1 += fmaxf(q, 0.f) * W_Q[o * DIMC + i];
    s2 += q * W_fc[o * DIMC + i];
  }
  qQ[b * DIMC + o] = s1 + b_Q[o];
  q2[b * DIMC + o] = s2 + b_fc[o];
}

// gather masked columns: Gf = bf16(relu(feat[:, :, idx])), Gp = bf16(pos[:, :, idx])
__global__ void gather_kernel(const float* __restrict__ feat, const float* __restrict__ pos,
                              const int* __restrict__ idx,
                              __bf16* __restrict__ Gf, __bf16* __restrict__ Gp) {
  size_t total = (size_t)BZC * DIMC * MC;
  size_t i = (size_t)blockIdx.x * blockDim.x + threadIdx.x;
  size_t st = (size_t)gridDim.x * blockDim.x;
  for (; i < total; i += st) {
    size_t j = i % MC, bc = i / MC;
    int s = idx[j];
    Gf[i] = f2bf(fmaxf(feat[bc * NNC + s], 0.f));
    Gp[i] = f2bf(pos[bc * NNC + s]);
  }
}

// ---------------------------------------------------------------------------
// Generic GEMM: C[o,col] = sum_i A[o,i]*B[i,col]  (A: 512x512 bf16 row-major,
// B: per-batch 512xNcols bf16). Block = 256 thr = 8 waves -> 64x32 tile.
// LDS: A row-major (TDM tile), B transposed (lBt[n][k]) so both WMMA fragments
// are contiguous b128 reads per lane.
// ---------------------------------------------------------------------------
enum EpiMode { EPI_MASK_F32 = 0, EPI_TANH_MASK_F32 = 1, EPI_ADD_BF16 = 2,
               EPI_RELU_F32_BF16 = 3, EPI_BF16_T = 4, EPI_RELU_MASK_F32 = 5 };

template <int EPI>
__global__ __launch_bounds__(256) void gemm512_bf16(
    const __bf16* __restrict__ A, const __bf16* __restrict__ Ball,
    const float* __restrict__ bias, const float* __restrict__ addv,
    float* __restrict__ outFall, __bf16* __restrict__ outBall, int Ncols) {
  const int bz = blockIdx.z;
  const __bf16* B = Ball + (size_t)bz * DIMC * Ncols;
  float* outF = outFall ? outFall + (size_t)bz * DIMC * Ncols : nullptr;
  __bf16* outB = outBall ? outBall + (size_t)bz * DIMC * Ncols : nullptr;
  const int n0 = blockIdx.x * 32, m0 = blockIdx.y * 64;
  __shared__ __align__(16) __bf16 lA[64 * 32];    // [oc][k]
  __shared__ __align__(16) __bf16 lBt[32 * 32];   // [n][k] (transposed)
  const int t = threadIdx.x;
  const int wave = t >> 5, lane = t & 31, hl = lane >> 4, l16 = lane & 15;
  const int wm = (wave >> 1) * 16, wn = (wave & 1) * 16;
  v8f acc = {};
  for (int kk = 0; kk < DIMC; kk += 32) {
#if USE_TDM
    if (wave == 0)
      tdm_load_tile_2d((unsigned)(size_t)(&lA[0]), A + (size_t)m0 * DIMC + kk,
                       DIMC, DIMC, 32, 64, DIMC);
#else
    {
      int row = t >> 2, cb = (t & 3) * 8;
      *(float4*)(lA + row * 32 + cb) = *(const float4*)(A + (size_t)(m0 + row) * DIMC + kk + cb);
    }
#endif
    {
      int rb = t >> 3, cb2 = (t & 7) * 4;   // row rb of B, 4 cols
      const __bf16* gb = B + (size_t)(kk + rb) * Ncols + n0 + cb2;
      __bf16 b0 = gb[0], b1 = gb[1], b2 = gb[2], b3 = gb[3];
      lBt[(cb2 + 0) * 32 + rb] = b0;
      lBt[(cb2 + 1) * 32 + rb] = b1;
      lBt[(cb2 + 2) * 32 + rb] = b2;
      lBt[(cb2 + 3) * 32 + rb] = b3;
      if (kk + 32 < DIMC)
        __builtin_prefetch(B + (size_t)(kk + 32 + rb) * Ncols + n0 + cb2, 0, 1);
    }
#if USE_TDM
    if (wave == 0) __builtin_amdgcn_s_wait_tensorcnt(0);
#endif
    __syncthreads();
    FragU ua, ub;
    const __bf16* ar = lA + (wm + l16) * 32;
    ua.f[0] = *(const float4*)(ar + hl * 8);
    ua.f[1] = *(const float4*)(ar + 16 + hl * 8);
    const __bf16* br = lBt + (wn + l16) * 32 + hl * 8;
    ub.f[0] = *(const float4*)(br);
    ub.f[1] = *(const float4*)(br + 16);
    acc = __builtin_amdgcn_wmma_f32_16x16x32_bf16(false, ua.v, false, ub.v, (short)0, acc, false, false);
    __syncthreads();
  }
  const int col = n0 + wn + l16;
  const float mk = ((col & 63) >= (col >> 6)) ? 1.f : 0.f;  // triu mask (NN layouts only)
#pragma unroll
  for (int r = 0; r < 8; ++r) {
    int row = m0 + wm + hl * 8 + r;
    float v = acc[r];
    if (bias) v += bias[row];
    if (EPI == EPI_ADD_BF16) v += addv[(size_t)bz * DIMC + row];
    size_t o = (size_t)row * Ncols + col;
    if (EPI == EPI_MASK_F32)            outF[o] = v * mk;
    else if (EPI == EPI_TANH_MASK_F32)  outF[o] = tanhf(v) * mk;
    else if (EPI == EPI_ADD_BF16)       outB[o] = f2bf(v);
    else if (EPI == EPI_RELU_F32_BF16) { v = fmaxf(v, 0.f); outF[o] = v; outB[o] = f2bf(v); }
    else if (EPI == EPI_BF16_T)         outB[(size_t)col * DIMC + row] = f2bf(v);  // (col, ch)
    else                                outF[o] = fmaxf(v, 0.f) * mk;
  }
}

// ---------------------------------------------------------------------------
// Flash attention over the 2080 masked positions. Block = 128 thr = 4 waves,
// wave == head. Q/K are position-major (b, j, 512) bf16 so A/B fragments are
// contiguous b128 loads; V stays channel-major (b, 512, M) which is already
// contiguous for the PV B-fragment. va out is channel-major f32.
// ---------------------------------------------------------------------------
__global__ __launch_bounds__(128) void attn_flash(
    const __bf16* __restrict__ Qt, const __bf16* __restrict__ Kt,
    const __bf16* __restrict__ Vall, float* __restrict__ vaAll) {
  const int bz = blockIdx.y;
  const int j0 = blockIdx.x * 16;
  const int wave = threadIdx.x >> 5;               // head index
  const int lane = threadIdx.x & 31, hl = lane >> 4, l16 = lane & 15;
  const __bf16* Vp = Vall + ((size_t)bz * DIMC + (size_t)wave * HDC) * MC;
  float* va = vaAll + ((size_t)bz * DIMC + (size_t)wave * HDC) * MC;

  // Q rows j0..j0+15, contiguous channel runs
  const __bf16* Qrow = Qt + ((size_t)bz * MC + j0 + l16) * DIMC + wave * HDC;
  v16bf aq[4];
#pragma unroll
  for (int ks = 0; ks < 4; ++ks) {
    FragU u;
    u.f[0] = *(const float4*)(Qrow + ks * 32 + hl * 8);
    u.f[1] = *(const float4*)(Qrow + ks * 32 + 16 + hl * 8);
    aq[ks] = u.v;
  }

  v8f acc[8]; float mrow[8], lrow[8];
  v8f zero = {};
#pragma unroll
  for (int i = 0; i < 8; ++i) { acc[i] = zero; mrow[i] = -3.0e38f; lrow[i] = 0.f; }

  __shared__ __align__(16) __bf16 lP[4][16 * 32];
  const float sc = 0.08838834764831845f;           // 1/sqrt(128)

  for (int c0 = 0; c0 < MC; c0 += 32) {
    const __bf16* Krow = Kt + ((size_t)bz * MC + c0 + l16) * DIMC + wave * HDC;
    v8f sl = {}, sr = {};
#pragma unroll
    for (int ks = 0; ks < 4; ++ks) {
      FragU ubl, ubr;
      ubl.f[0] = *(const float4*)(Krow + ks * 32 + hl * 8);
      ubl.f[1] = *(const float4*)(Krow + ks * 32 + hl * 8 + 16);
      ubr.f[0] = *(const float4*)(Krow + 16 * DIMC + ks * 32 + hl * 8);
      ubr.f[1] = *(const float4*)(Krow + 16 * DIMC + ks * 32 + hl * 8 + 16);
      sl = __builtin_amdgcn_wmma_f32_16x16x32_bf16(false, aq[ks], false, ubl.v, (short)0, sl, false, false);
      sr = __builtin_amdgcn_wmma_f32_16x16x32_bf16(false, aq[ks], false, ubr.v, (short)0, sr, false, false);
    }
    float pl[8], pr[8];
#pragma unroll
    for (int r = 0; r < 8; ++r) {
      float a = sl[r] * sc, b2 = sr[r] * sc;
      float mx = fmaxf(a, b2);
#pragma unroll
      for (int d = 1; d < 16; d <<= 1) mx = fmaxf(mx, __shfl_xor(mx, d, 32));
      float mnew = fmaxf(mrow[r], mx);
      float co = __expf(mrow[r] - mnew);
      pl[r] = __expf(a - mnew); pr[r] = __expf(b2 - mnew);
      float rs = pl[r] + pr[r];
#pragma unroll
      for (int d = 1; d < 16; d <<= 1) rs += __shfl_xor(rs, d, 32);
      lrow[r] = lrow[r] * co + rs;
      mrow[r] = mnew;
#pragma unroll
      for (int tt = 0; tt < 8; ++tt) acc[tt][r] *= co;
    }
    __syncthreads();
#pragma unroll
    for (int r = 0; r < 8; ++r) {
      lP[wave][(hl * 8 + r) * 32 + l16] = f2bf(pl[r]);
      lP[wave][(hl * 8 + r) * 32 + 16 + l16] = f2bf(pr[r]);
    }
    __syncthreads();
    FragU up;
    const __bf16* pr0 = &lP[wave][l16 * 32];
    up.f[0] = *(const float4*)(pr0 + hl * 8);
    up.f[1] = *(const float4*)(pr0 + 16 + hl * 8);
#pragma unroll
    for (int tt = 0; tt < 8; ++tt) {
      FragU uv;
      const __bf16* Vrow = Vp + (size_t)(tt * 16 + l16) * MC + c0 + hl * 8;
      uv.f[0] = *(const float4*)(Vrow);
      uv.f[1] = *(const float4*)(Vrow + 16);
      acc[tt] = __builtin_amdgcn_wmma_f32_16x16x32_bf16(false, up.v, false, uv.v, (short)0, acc[tt], false, false);
    }
  }
#pragma unroll
  for (int tt = 0; tt < 8; ++tt)
#pragma unroll
    for (int r = 0; r < 8; ++r)
      va[(size_t)(tt * 16 + l16) * MC + j0 + hl * 8 + r] = acc[tt][r] / lrow[r];
}

// ---------------------------------------------------------------------------
// comb + q2 scale + channel L2-normalize + mask -> bf16 activation for convs
// ---------------------------------------------------------------------------
__global__ __launch_bounds__(256) void comb_norm_kernel(
    const float* __restrict__ feat, const float* __restrict__ va,
    const float* __restrict__ Vf, const float* __restrict__ q2,
    __bf16* __restrict__ xb) {
  const int bs = blockIdx.x;
  const int b = bs >> 12, s = bs & 4095;
  const int n = s >> 6, m = s & 63;
  const bool msk = (m >= n);
  const int j = msk ? (n * 64 - (n * (n - 1)) / 2 + (m - n)) : 0;
  const int t = threadIdx.x;
  __shared__ float red[256];
  const int c0 = t, c1 = t + 256;
  const size_t bc0 = (size_t)b * DIMC + c0, bc1 = (size_t)b * DIMC + c1;
  float comb0 = feat[bc0 * NNC + s], comb1 = feat[bc1 * NNC + s];
  if (msk) {
    comb0 += fmaxf(va[bc0 * MC + j] + Vf[bc0 * MC + j], 0.f);
    comb1 += fmaxf(va[bc1 * MC + j] + Vf[bc1 * MC + j], 0.f);
  }
  float x0 = q2[b * DIMC + c0] * comb0;
  float x1 = q2[b * DIMC + c1] * comb1;
  red[t] = x0 * x0 + x1 * x1;
  __syncthreads();
  for (int off = 128; off > 0; off >>= 1) { if (t < off) red[t] += red[t + off]; __syncthreads(); }
  float scale = msk ? 1.f / fmaxf(sqrtf(red[0]), 1e-12f) : 0.f;
  xb[bc0 * NNC + s] = f2bf(x0 * scale);
  xb[bc1 * NNC + s] = f2bf(x1 * scale);
}

// ---------------------------------------------------------------------------
// 5x5 conv as implicit GEMM (K = 25 taps x 512 ch), tap-major weights Wt so
// the A tile is a dense row-major 64x32 block (TDM / b128 loads).
// ---------------------------------------------------------------------------
__global__ __launch_bounds__(256) void conv5x5_wmma(
    const __bf16* __restrict__ Xall, const __bf16* __restrict__ Wt,
    const float* __restrict__ bias, __bf16* __restrict__ Yall) {
  const int bz = blockIdx.z;
  const __bf16* X = Xall + (size_t)bz * DIMC * NNC;
  __bf16* Y = Yall + (size_t)bz * DIMC * NNC;
  const int s0 = blockIdx.x * 32;
  const int n = s0 >> 6, m0 = s0 & 63;
  const int ocb = blockIdx.y * 64;
  __shared__ __align__(16) __bf16 lA[64 * 32];    // [oc][ci]
  __shared__ __align__(16) __bf16 lBt[32 * 32];   // [scol][ci] (transposed)
  const int t = threadIdx.x;
  const int wave = t >> 5, lane = t & 31, hl = lane >> 4, l16 = lane & 15;
  const int wm = (wave >> 1) * 16, wn = (wave & 1) * 16;
  const __bf16 zb = f2bf(0.f);
  v8f acc = {};
  for (int tap = 0; tap < 25; ++tap) {
    const int dn = tap / 5 - 2, dm = tap % 5 - 2;
    const int n2 = n + dn;
    const bool rowOK = (n2 >= 0) && (n2 < 64);
    for (int c0 = 0; c0 < DIMC; c0 += 32) {
#if USE_TDM
      if (wave == 0)
        tdm_load_tile_2d((unsigned)(size_t)(&lA[0]),
                         Wt + ((size_t)tap * DIMC + ocb) * DIMC + c0,
                         DIMC, DIMC, 32, 64, DIMC);
#else
      {
        int row = t >> 2, cb = (t & 3) * 8;
        *(float4*)(lA + row * 32 + cb) =
            *(const float4*)(Wt + ((size_t)tap * DIMC + ocb + row) * DIMC + c0 + cb);
      }
#endif
      {
        int k = t >> 3, sb = (t & 7) * 4;
#pragma unroll
        for (int r = 0; r < 4; ++r) {
          int sc = sb + r;
          int mm = m0 + sc + dm;
          __bf16 v = zb;
          if (rowOK && mm >= 0 && mm < 64)
            v = X[(size_t)(c0 + k) * NNC + n2 * 64 + mm];
          lBt[sc * 32 + k] = v;
        }
      }
#if USE_TDM
      if (wave == 0) __builtin_amdgcn_s_wait_tensorcnt(0);
#endif
      __syncthreads();
      FragU ua, ub;
      const __bf16* ar = lA + (wm + l16) * 32;
      ua.f[0] = *(const float4*)(ar + hl * 8);
      ua.f[1] = *(const float4*)(ar + 16 + hl * 8);
      const __bf16* br = lBt + (wn + l16) * 32 + hl * 8;
      ub.f[0] = *(const float4*)(br);
      ub.f[1] = *(const float4*)(br + 16);
      acc = __builtin_amdgcn_wmma_f32_16x16x32_bf16(false, ua.v, false, ub.v, (short)0, acc, false, false);
      __syncthreads();
    }
  }
  const int m = m0 + wn + l16;
  const float mk = (m >= n) ? 1.f : 0.f;
#pragma unroll
  for (int r = 0; r < 8; ++r) {
    int oc = ocb + wm + hl * 8 + r;
    float v = fmaxf(acc[r] + bias[oc], 0.f) * mk;
    Y[(size_t)oc * NNC + n * 64 + m] = f2bf(v);
  }
}

// scores2d[b,n,m] = sum_c pred_w[c] * (m3[b,c,s] + R[b,c,s])
__global__ __launch_bounds__(256) void score_kernel(
    const __bf16* __restrict__ m3, const float* __restrict__ R,
    const float* __restrict__ pw, float* __restrict__ out) {
  const int bs = blockIdx.x;
  const int b = bs >> 12, s = bs & 4095;
  const int t = threadIdx.x;
  __shared__ float red[256];
  const size_t bc0 = (size_t)b * DIMC + t, bc1 = (size_t)b * DIMC + t + 256;
  float p = pw[t]       * (bf2f(m3[bc0 * NNC + s]) + R[bc0 * NNC + s]) +
            pw[t + 256] * (bf2f(m3[bc1 * NNC + s]) + R[bc1 * NNC + s]);
  red[t] = p;
  __syncthreads();
  for (int off = 128; off > 0; off >>= 1) { if (t < off) red[t] += red[t + off]; __syncthreads(); }
  if (t == 0) out[(size_t)b * NNC + s] = red[0];
}

// ---------------------------------------------------------------------------
extern "C" void kernel_launch(void* const* d_in, const int* in_sizes, int n_in,
                              void* d_out, int out_size, void* d_ws, size_t ws_size,
                              hipStream_t stream) {
  (void)in_sizes; (void)n_in; (void)out_size; (void)ws_size;
  const float* qry    = (const float*)d_in[0];
  const float* map2d  = (const float*)d_in[1];
  /* d_in[2] = mask_f: unused, mask recomputed analytically (m >= n) */
  const int*   idx    = (const int*)d_in[3];
  const float* W_conv = (const float*)d_in[4];  const float* b_conv = (const float*)d_in[5];
  const float* W_pos  = (const float*)d_in[6];  const float* b_pos  = (const float*)d_in[7];
  const float* W_res  = (const float*)d_in[8];  const float* b_res  = (const float*)d_in[9];
  const float* W_c1   = (const float*)d_in[10]; const float* b_c1   = (const float*)d_in[11];
  const float* W_c2   = (const float*)d_in[12]; const float* b_c2   = (const float*)d_in[13];
  const float* W_c3   = (const float*)d_in[14]; const float* b_c3   = (const float*)d_in[15];
  const float* W_fc   = (const float*)d_in[16]; const float* b_fc   = (const float*)d_in[17];
  const float* W_Q    = (const float*)d_in[18]; const float* b_Q    = (const float*)d_in[19];
  const float* W_M    = (const float*)d_in[20]; const float* b_M    = (const float*)d_in[21];
  const float* pred_w = (const float*)d_in[22];
  const float* W_val  = (const float*)d_in[23]; const float* b_val  = (const float*)d_in[24];
  const float* W_key  = (const float*)d_in[25];
  const float* W_qry  = (const float*)d_in[26];

  float* outScores = (float*)d_out;                         // (8,64,64)
  float* outPos    = outScores + (size_t)BZC * NNC;         // (8,512,64,64)
  float* outFeat   = outPos + (size_t)BZC * DIMC * NNC;     // (8,512,64,64)

  char* w = (char*)d_ws;
  auto take = [&](size_t bytes) -> char* {
    char* p = w; w += (bytes + 255) & ~(size_t)255; return p;
  };
  const size_t NB  = (size_t)BZC * DIMC * NNC;  // full-grid tensor elems
  const size_t MB  = (size_t)BZC * DIMC * MC;   // masked tensor elems
  __bf16* map2d_bf = (__bf16*)take(NB * 2);
  __bf16* Wconv_bf = (__bf16*)take((size_t)DIMC * DIMC * 2);
  __bf16* Wpos_bf  = (__bf16*)take((size_t)DIMC * DIMC * 2);
  __bf16* Wres_bf  = (__bf16*)take((size_t)DIMC * DIMC * 2);
  __bf16* WM_bf    = (__bf16*)take((size_t)DIMC * DIMC * 2);
  __bf16* Wval_bf  = (__bf16*)take((size_t)DIMC * DIMC * 2);
  __bf16* Wkey_bf  = (__bf16*)take((size_t)DIMC * DIMC * 2);
  __bf16* Wqry_bf  = (__bf16*)take((size_t)DIMC * DIMC * 2);
  __bf16* Wt1      = (__bf16*)take((size_t)25 * DIMC * DIMC * 2);
  __bf16* Wt2      = (__bf16*)take((size_t)25 * DIMC * DIMC * 2);
  __bf16* Wt3      = (__bf16*)take((size_t)25 * DIMC * DIMC * 2);
  float*  qQ       = (float*)take((size_t)BZC * DIMC * 4);
  float*  q2       = (float*)take((size_t)BZC * DIMC * 4);
  __bf16* Gf       = (__bf16*)take(MB * 2);
  __bf16* Gp       = (__bf16*)take(MB * 2);
  __bf16* qg_bf    = (__bf16*)take(MB * 2);
  __bf16* Qt       = (__bf16*)take(MB * 2);   // position-major (b, j, 512)
  __bf16* Kt       = (__bf16*)take(MB * 2);   // position-major (b, j, 512)
  __bf16* Vb       = (__bf16*)take(MB * 2);   // channel-major  (b, 512, M)
  float*  Vf       = (float*)take(MB * 4);
  float*  va       = (float*)take(MB * 4);
  __bf16* xb       = (__bf16*)take(NB * 2);
  __bf16* act1     = (__bf16*)take(NB * 2);
  __bf16* act2     = (__bf16*)take(NB * 2);
  __bf16* act3     = (__bf16*)take(NB * 2);
  float*  Rbuf     = (float*)take(NB * 4);

  auto cvt = [&](const float* src, __bf16* dst, size_t nElems) {
    size_t blocks = (nElems + 255) / 256; if (blocks > 8192) blocks = 8192;
    cvt_bf16<<<dim3((unsigned)blocks), dim3(256), 0, stream>>>(src, dst, nElems);
  };
  cvt(map2d, map2d_bf, NB);
  cvt(W_conv, Wconv_bf, (size_t)DIMC * DIMC);
  cvt(W_pos,  Wpos_bf,  (size_t)DIMC * DIMC);
  cvt(W_res,  Wres_bf,  (size_t)DIMC * DIMC);
  cvt(W_M,    WM_bf,    (size_t)DIMC * DIMC);
  cvt(W_val,  Wval_bf,  (size_t)DIMC * DIMC);
  cvt(W_key,  Wkey_bf,  (size_t)DIMC * DIMC);
  cvt(W_qry,  Wqry_bf,  (size_t)DIMC * DIMC);
  repack_convw<<<dim3(8192), dim3(256), 0, stream>>>(W_c1, Wt1);
  repack_convw<<<dim3(8192), dim3(256), 0, stream>>>(W_c2, Wt2);
  repack_convw<<<dim3(8192), dim3(256), 0, stream>>>(W_c3, Wt3);

  qvec_kernel<<<dim3(BZC), dim3(DIMC), 0, stream>>>(qry, W_Q, b_Q, W_fc, b_fc, qQ, q2);

  const dim3 g4096(NNC / 32, DIMC / 64, BZC);
  const dim3 g2080(MC / 32, DIMC / 64, BZC);
  const dim3 blk(256);

  // feat = conv1x1(map2d, W_conv) * mask ; pos = tanh(conv1x1(map2d, W_pos)) * mask
  gemm512_bf16<EPI_MASK_F32><<<g4096, blk, 0, stream>>>(Wconv_bf, map2d_bf, b_conv, nullptr, outFeat, nullptr, NNC);
  gemm512_bf16<EPI_TANH_MASK_F32><<<g4096, blk, 0, stream>>>(Wpos_bf, map2d_bf, b_pos, nullptr, outPos, nullptr, NNC);

  {
    size_t blocks = (MB + 255) / 256; if (blocks > 16384) blocks = 16384;
    gather_kernel<<<dim3((unsigned)blocks), blk, 0, stream>>>(outFeat, outPos, idx, Gf, Gp);
  }

  // qg = W_M @ relu(feat_gathered) + b_M + qQ   (bf16, channel-major)
  gemm512_bf16<EPI_ADD_BF16><<<g2080, blk, 0, stream>>>(WM_bf, Gf, b_M, qQ, nullptr, qg_bf, MC);
  // value = relu(W_val @ pe + b_val)  (f32 + bf16, channel-major)
  gemm512_bf16<EPI_RELU_F32_BF16><<<g2080, blk, 0, stream>>>(Wval_bf, Gp, b_val, nullptr, Vf, Vb, MC);
  // key / qv -> position-major (b, j, 512) for contiguous attention fragments
  gemm512_bf16<EPI_BF16_T><<<g2080, blk, 0, stream>>>(Wkey_bf, Gp, nullptr, nullptr, nullptr, Kt, MC);
  gemm512_bf16<EPI_BF16_T><<<g2080, blk, 0, stream>>>(Wqry_bf, qg_bf, nullptr, nullptr, nullptr, Qt, MC);

  attn_flash<<<dim3(MC / 16, BZC), dim3(128), 0, stream>>>(Qt, Kt, Vb, va);

  comb_norm_kernel<<<dim3(BZC * NNC), blk, 0, stream>>>(outFeat, va, Vf, q2, xb);

  // residual branch: relu(conv1x1(map2d_f, W_res) + b_res) * mask  (f32)
  gemm512_bf16<EPI_RELU_MASK_F32><<<g4096, blk, 0, stream>>>(Wres_bf, xb, b_res, nullptr, Rbuf, nullptr, NNC);

  const dim3 gconv(NNC / 32, DIMC / 64, BZC);
  conv5x5_wmma<<<gconv, blk, 0, stream>>>(xb,   Wt1, b_c1, act1);
  conv5x5_wmma<<<gconv, blk, 0, stream>>>(act1, Wt2, b_c2, act2);
  conv5x5_wmma<<<gconv, blk, 0, stream>>>(act2, Wt3, b_c3, act3);

  score_kernel<<<dim3(BZC * NNC), blk, 0, stream>>>(act3, Rbuf, pred_w, outScores);
}
